// ChunkModule_17214228922854
// MI455X (gfx1250) — compile-verified, matrix operand order checked by reading, not run on hardware
//
#include <hip/hip_runtime.h>
#include <math.h>

// ---------------------------------------------------------------------------
// Types / helpers
// ---------------------------------------------------------------------------
typedef __attribute__((ext_vector_type(16))) unsigned short v16u;
typedef __attribute__((ext_vector_type(8)))  unsigned short v8u;
typedef __attribute__((ext_vector_type(16))) __bf16        v16bf;
typedef __attribute__((ext_vector_type(8)))  float         v8f;

#define DEV __device__ __forceinline__

DEV unsigned short bf16_rn(float f) {
  unsigned int u = __float_as_uint(f);
  u += 0x7FFFu + ((u >> 16) & 1u);   // round-to-nearest-even
  return (unsigned short)(u >> 16);
}

DEV float gelu_f(float x) { return 0.5f * x * (1.0f + erff(x * 0.70710678118654752440f)); }
DEV float silu_f(float x) { return x / (1.0f + expf(-x)); }
DEV float act_f(float x, int act) { return act == 1 ? gelu_f(x) : x; }

DEV float block_reduce_sum(float v, float* sm) {
  int tid = threadIdx.x;
  sm[tid] = v;
  __syncthreads();
  for (int s = 128; s > 0; s >>= 1) {
    if (tid < s) sm[tid] += sm[tid + s];
    __syncthreads();
  }
  float r = sm[0];
  __syncthreads();
  return r;
}

// ---------------------------------------------------------------------------
// CDNA5 async global->LDS copy (ASYNCcnt) + waits, per CDNA5 ISA
// ---------------------------------------------------------------------------
DEV void async_cp16(unsigned short* dst_lds, const unsigned short* src) {
  unsigned loff =
      (unsigned)(unsigned long long)(__attribute__((address_space(3))) unsigned short*)dst_lds;
  asm volatile("global_load_async_to_lds_b128 %0, %1, off"
               :: "v"(loff), "v"((unsigned long long)(size_t)src)
               : "memory");
}
DEV void wait_async3() { asm volatile("s_wait_asynccnt 0x3" ::: "memory"); }
DEV void wait_async0() { asm volatile("s_wait_asynccnt 0x0" ::: "memory"); }

// ---------------------------------------------------------------------------
// WMMA plumbing
// ---------------------------------------------------------------------------
DEV v8f wmma_bf16(v16u a, v16u b, v8f c) {
  return __builtin_amdgcn_wmma_f32_16x16x32_bf16(
      false, __builtin_bit_cast(v16bf, a),
      false, __builtin_bit_cast(v16bf, b),
      (short)0, c, false, false);
}

// A-frag (16x32): lane L<16 -> row L, K {0..7,16..23}; lane L>=16 -> row L-16,
// K {8..15,24..31}. Tile rows in LDS with stride 40 halves (80B: conflict-free).
#define LDS_STRIDE 40
DEV v16u frag_lds(const unsigned short* tile, int row, int khalf) {
  const unsigned short* p = tile + row * LDS_STRIDE + khalf;
  v8u lo = *(const v8u*)p;          // ds_load_b128
  v8u hi = *(const v8u*)(p + 16);   // ds_load_b128
  return __builtin_shufflevector(lo, hi, 0, 1, 2, 3, 4, 5, 6, 7,
                                 8, 9, 10, 11, 12, 13, 14, 15);
}

DEV void store_tile(v8f c, float* __restrict__ OUT, int row, int col,
                    const float* __restrict__ bias, int N, int ldo, int act) {
  if (col >= N) return;
  float bv = bias ? bias[col] : 0.0f;
#pragma unroll
  for (int r = 0; r < 8; ++r) {
    OUT[(size_t)(row + r) * ldo + col] = act_f(c[r] + bv, act);
  }
}

// ---------------------------------------------------------------------------
// Fused GEMM / multi-tap conv1d kernel with LDS double buffering.
//   OUT[M,N] = act( sum_u A_shift(u-pad) @ W[u]^T + bias )
// A : bf16 [M,K] row-major, W : bf16 [ntaps, N, K]
// lbmask = tokensPerBatch-1 (power of two); row shifts masked at batch edges.
// block = 256 threads (8 waves, 4x2) -> 128(M) x 64(N) tile; wave tile 32x32.
// M multiple of 128 at all call sites (4096 / 8192); K multiple of 32.
// ---------------------------------------------------------------------------
__global__ __launch_bounds__(256) void gemm_conv_wmma(
    const unsigned short* __restrict__ A, const unsigned short* __restrict__ W,
    const float* __restrict__ bias, float* __restrict__ OUT,
    int M, int N, int K, int lbmask, int ntaps, int pad, int act, int ldo)
{
  __shared__ __align__(16) unsigned short sA[2][128 * LDS_STRIDE];
  __shared__ __align__(16) unsigned short sB[2][64 * LDS_STRIDE];
  __shared__ __align__(16) unsigned short sDummy[8];

  const int tid = threadIdx.x;
  const int bx128 = blockIdx.x * 128;
  const int by64  = blockIdx.y * 64;
  const int ksteps = K >> 5;
  const int nsteps = ntaps * ksteps;

  // Stage tile for step -> buffer bi. Always issues exactly 3 async loads per
  // wave (invalid chunks load a dummy slot from a safe address and zero the
  // real slot via ds_store) so s_wait_asynccnt 3 deterministically drains the
  // previous stage.
  auto stage = [&](int step, int bi) {
    int u = step / ksteps;
    int k0 = (step - u * ksteps) << 5;
    int s = u - pad;
    // A tile: 128 rows x 32 halves = 512 x 16B chunks, 2 per thread
#pragma unroll
    for (int h2 = 0; h2 < 2; ++h2) {
      int c = tid + (h2 << 8);
      int row = c >> 2, cc = (c & 3) << 3;
      unsigned short* dst = &sA[bi][row * LDS_STRIDE + cc];
      int ga = bx128 + row;
      int t = ga & lbmask;
      bool v = (unsigned)(t + s) <= (unsigned)lbmask;
      const unsigned short* src = v ? (A + (size_t)(ga + s) * K + k0 + cc) : A;
      async_cp16(v ? dst : sDummy, src);
      if (!v) { v8u zz = {}; *(v8u*)dst = zz; }
    }
    // B tile: 64 rows x 32 halves = 256 x 16B chunks, 1 per thread
    {
      int row = tid >> 2, cc = (tid & 3) << 3;
      unsigned short* dst = &sB[bi][row * LDS_STRIDE + cc];
      int n = by64 + row;
      bool v = n < N;
      const unsigned short* src = v ? (W + ((size_t)u * N + n) * K + k0 + cc) : W;
      async_cp16(v ? dst : sDummy, src);
      if (!v) { v8u zz = {}; *(v8u*)dst = zz; }
    }
  };

  const int wave = tid >> 5, lane = tid & 31;
  const int wm = wave & 3, wn = wave >> 2;
  const int lrow = lane & 15;
  const int khalf = (lane >> 4) << 3;
  const int arow = wm * 32 + lrow;
  const int brow = wn * 32 + lrow;

  v8f c00 = {}, c01 = {}, c10 = {}, c11 = {};

  stage(0, 0);
  for (int i = 0; i < nsteps; ++i) {
    int bi = i & 1;
    if (i + 1 < nsteps) {
      stage(i + 1, bi ^ 1);
      wait_async3();           // previous stage's 3 asyncs complete
    } else {
      wait_async0();
    }
    __syncthreads();           // all waves' stage writes visible
    v16u a0 = frag_lds(&sA[bi][0], arow,      khalf);
    v16u a1 = frag_lds(&sA[bi][0], arow + 16, khalf);
    v16u b0 = frag_lds(&sB[bi][0], brow,      khalf);
    v16u b1 = frag_lds(&sB[bi][0], brow + 16, khalf);
    c00 = wmma_bf16(a0, b0, c00);
    c01 = wmma_bf16(a0, b1, c01);
    c10 = wmma_bf16(a1, b0, c10);
    c11 = wmma_bf16(a1, b1, c11);
    __syncthreads();           // safe to overwrite buffer bi next round
  }

  const int lcol = lane & 15;
  const int radd = (lane >> 4) << 3;
  const int m0 = bx128 + wm * 32;
  const int n0 = by64 + wn * 32;
  store_tile(c00, OUT, m0 + radd,      n0 + lcol,      bias, N, ldo, act);
  store_tile(c01, OUT, m0 + radd,      n0 + 16 + lcol, bias, N, ldo, act);
  store_tile(c10, OUT, m0 + 16 + radd, n0 + lcol,      bias, N, ldo, act);
  store_tile(c11, OUT, m0 + 16 + radd, n0 + 16 + lcol, bias, N, ldo, act);
}

// ---------------------------------------------------------------------------
// fp32 -> bf16 mirror (activations / weights for WMMA)
// ---------------------------------------------------------------------------
__global__ void cvt_bf16_kernel(const float* __restrict__ in,
                                unsigned short* __restrict__ out, size_t n) {
  size_t i = (size_t)blockIdx.x * blockDim.x + threadIdx.x;
  if (i < n) out[i] = bf16_rn(in[i]);
}

// conv weight (O,I,k) -> bf16 [k][O][I]
__global__ void repack_conv_bf16(const float* __restrict__ w,
                                 unsigned short* __restrict__ out,
                                 int O, int I, int k) {
  size_t idx = (size_t)blockIdx.x * blockDim.x + threadIdx.x;
  size_t total = (size_t)O * I * k;
  if (idx >= total) return;
  int u = (int)(idx % k);
  size_t oi = idx / k;
  int i = (int)(oi % I);
  int o = (int)(oi / I);
  out[((size_t)u * O + o) * I + i] = bf16_rn(w[idx]);
}

// ---------------------------------------------------------------------------
// LayerNorm per token (cols = 512), optional residual input
// ---------------------------------------------------------------------------
__global__ void ln_kernel(const float* __restrict__ a, const float* __restrict__ res,
                          const float* __restrict__ w, const float* __restrict__ bb,
                          float* __restrict__ out, int cols)
{
  __shared__ float sm[256];
  size_t g = blockIdx.x;
  int per = cols >> 8;
  float v[4];
  float s = 0.0f;
  for (int e = 0; e < per; ++e) {
    int col = threadIdx.x + (e << 8);
    float x = a[g * cols + col];
    if (res) x += res[g * cols + col];
    v[e] = x; s += x;
  }
  float mean = block_reduce_sum(s, sm) / (float)cols;
  float s2 = 0.0f;
  for (int e = 0; e < per; ++e) { float d = v[e] - mean; s2 += d * d; }
  float var = block_reduce_sum(s2, sm) / (float)cols;
  float inv = rsqrtf(var + 1e-5f);
  for (int e = 0; e < per; ++e) {
    int col = threadIdx.x + (e << 8);
    out[g * cols + col] = (v[e] - mean) * inv * w[col] + bb[col];
  }
}

// ---------------------------------------------------------------------------
// Depthwise causal conv (d_conv=4) + bias + SiLU over the xBC slice of zxbcdt.
// rev=1 flips causality (equivalent to flip -> conv -> flip).
// ---------------------------------------------------------------------------
__global__ void dwconv_silu_kernel(const float* __restrict__ zx, int zx_ld, int off,
                                   const float* __restrict__ cw, const float* __restrict__ cb,
                                   float* __restrict__ out, int cdim, int Btok, int nb, int rev)
{
  size_t idx = (size_t)blockIdx.x * blockDim.x + threadIdx.x;
  size_t total = (size_t)nb * Btok * cdim;
  if (idx >= total) return;
  int c = (int)(idx % cdim);
  size_t gg = idx / cdim;
  int t = (int)(gg % Btok);
  int b = (int)(gg / Btok);
  float acc = cb[c];
#pragma unroll
  for (int j = 0; j < 4; ++j) {
    int src = t + (rev ? (3 - j) : (j - 3));
    if ((unsigned)src < (unsigned)Btok)
      acc += cw[c * 4 + j] * zx[((size_t)b * Btok + src) * zx_ld + off + c];
  }
  out[gg * (size_t)cdim + c] = silu_f(acc);
}

// ---------------------------------------------------------------------------
// Selective SSM scan. One block per (batch, head); one thread per P-channel.
// State [DSTATE] fully register-resident. Folds the D*x skip term into y.
// ---------------------------------------------------------------------------
template <int DS>
__global__ void scan_kernel(const float* __restrict__ zx, int zx_ld, int dt_off,
                            const float* __restrict__ xc, int xc_ld,
                            const float* __restrict__ dt_bias, const float* __restrict__ A_log,
                            const float* __restrict__ Dp,
                            float* __restrict__ yout, int y_ld,
                            int Btok, int H, int rev)
{
  int h = blockIdx.x % H, b = blockIdx.x / H;
  int P = blockDim.x, p = threadIdx.x;
  float aneg = -expf(A_log[h]);
  float tb = dt_bias[h], Dh = Dp[h];
  float st[DS];
#pragma unroll
  for (int n = 0; n < DS; ++n) st[n] = 0.0f;
  for (int i = 0; i < Btok; ++i) {
    int t = rev ? (Btok - 1 - i) : i;
    size_t g = (size_t)b * Btok + t;
    float dv = zx[g * zx_ld + dt_off + h] + tb;
    float dt = dv > 20.0f ? dv : log1pf(expf(dv));   // softplus
    float da = expf(dt * aneg);
    float xp = xc[g * xc_ld + h * P + p];
    const float* Bv = xc + g * xc_ld + 1024;
    const float* Cv = Bv + DS;
    float coef = dt * xp;
    float yv = Dh * xp;
#pragma unroll
    for (int n = 0; n < DS; ++n) {
      float sn = st[n] * da + coef * Bv[n];
      st[n] = sn;
      yv += sn * Cv[n];
    }
    yout[g * y_ld + h * P + p] = yv;
  }
}

// ---------------------------------------------------------------------------
// Gated RMSNorm: out = rmsnorm(y * silu(z)) * norm_w, cols = 1024
// ---------------------------------------------------------------------------
__global__ void gate_rms_kernel(const float* __restrict__ y, const float* __restrict__ zx,
                                int zx_ld, const float* __restrict__ nw,
                                float* __restrict__ out)
{
  __shared__ float sm[256];
  size_t g = blockIdx.x;
  float v[4];
  float ss = 0.0f;
#pragma unroll
  for (int e = 0; e < 4; ++e) {
    int col = threadIdx.x + (e << 8);
    float z = zx[g * (size_t)zx_ld + col];
    float val = y[g * 1024 + col] * silu_f(z);
    v[e] = val; ss += val * val;
  }
  float ms = block_reduce_sum(ss, sm) * (1.0f / 1024.0f);
  float inv = rsqrtf(ms + 1e-5f);
#pragma unroll
  for (int e = 0; e < 4; ++e) {
    int col = threadIdx.x + (e << 8);
    out[g * 1024 + col] = v[e] * inv * nw[col];
  }
}

// ---------------------------------------------------------------------------
// Elementwise utilities
// ---------------------------------------------------------------------------
__global__ void affine_kernel(const float* __restrict__ in, const float* __restrict__ g,
                              const float* __restrict__ b, float* __restrict__ out,
                              size_t total, int cols, int act) {
  size_t i = (size_t)blockIdx.x * blockDim.x + threadIdx.x;
  if (i >= total) return;
  int c = (int)(i % cols);
  float v = in[i] * (g[c] * 0.99999500003749971875f) + b[c];   // BN_INV = 1/sqrt(1+1e-5)
  out[i] = act ? gelu_f(v) : v;
}

__global__ void add_kernel(const float* __restrict__ a, const float* __restrict__ b,
                           float* __restrict__ out, size_t n) {
  size_t i = (size_t)blockIdx.x * blockDim.x + threadIdx.x;
  if (i < n) out[i] = a[i] + b[i];
}

// batch-strided row block copy: dst[(b*dstS + r)*cols + c] = src[(b*srcS + r)*cols + c]
__global__ void copy_block_kernel(const float* __restrict__ src, float* __restrict__ dst,
                                  int rowsPerBatch, int cols, int nb,
                                  int srcS, int dstS) {
  size_t idx = (size_t)blockIdx.x * blockDim.x + threadIdx.x;
  size_t total = (size_t)nb * rowsPerBatch * cols;
  if (idx >= total) return;
  int c = (int)(idx % cols);
  size_t r = idx / cols;
  int b = (int)(r / rowsPerBatch);
  int rr = (int)(r % rowsPerBatch);
  dst[((size_t)b * dstS + rr) * cols + c] = src[((size_t)b * srcS + rr) * cols + c];
}

__global__ void concat_kernel(const float* __restrict__ a, const float* __restrict__ b,
                              float* __restrict__ out, size_t rows) {
  size_t idx = (size_t)blockIdx.x * blockDim.x + threadIdx.x;
  size_t total = rows * 1024;
  if (idx >= total) return;
  int col = (int)(idx % 1024);
  size_t r = idx / 1024;
  out[idx] = (col < 512) ? a[r * 512 + col] : b[r * 512 + (col - 512)];
}

// ---------------------------------------------------------------------------
// Host orchestration
// ---------------------------------------------------------------------------
extern "C" void kernel_launch(void* const* d_in, const int* in_sizes, int n_in,
                              void* d_out, int out_size, void* d_ws, size_t ws_size,
                              hipStream_t stream)
{
  (void)in_sizes; (void)n_in; (void)out_size; (void)ws_size;
  const float* const* in = (const float* const*)d_in;
  const int B = 2, L = 2048;
  const int NT = B * L;                       // 4096 tokens

  float* ws = (float*)d_ws;
  size_t off = 0;
  auto alloc = [&](size_t n) { float* p = ws + off; off += (n + 3) & ~(size_t)3; return p; };

  float* XA0  = alloc((size_t)NT * 512);      // bimamba output (persist)
  float* SKIP = alloc((size_t)NT * 512);      // convblock2 output (persist)
  float* CUR  = alloc((size_t)NT * 512);
  float* NXT  = alloc((size_t)NT * 512);
  float* T0   = alloc((size_t)NT * 512);
  float* T1   = alloc((size_t)NT * 512);
  float* T2   = alloc((size_t)NT * 512);
  float* BI   = alloc((size_t)NT * 1024);     // [fwd | bwd] mamba outputs
  float* HID  = alloc((size_t)NT * 2048);     // FFN hidden
  float* CAT  = alloc((size_t)2 * NT * 512);  // stackmamba concat input / out_proj output
  float* ZX   = alloc((size_t)2 * NT * 2184); // zxbcdt
  float* XC   = alloc((size_t)2 * NT * 1152); // conv+silu(xBC)
  float* YS   = alloc((size_t)2 * NT * 1024); // scan outputs
  float* GT   = alloc((size_t)2 * NT * 1024); // gated rmsnorm outputs
  unsigned short* AB = (unsigned short*)alloc((size_t)4194304);  // bf16 activation mirror
  unsigned short* WB = (unsigned short*)alloc((size_t)1966096);  // bf16 weight mirror

  auto cvt = [&](const float* src, unsigned short* dst, size_t n) {
    cvt_bf16_kernel<<<(unsigned)((n + 255) / 256), 256, 0, stream>>>(src, dst, n);
  };

  // GEMM: out[M,N](ldo) = act(A[M,K] @ W[N,K]^T + bias)
  auto gemm = [&](const float* A, const float* W, const float* bias, float* out,
                  int M, int N, int K, int act, int ldo) {
    cvt(A, AB, (size_t)M * K);
    cvt(W, WB, (size_t)N * K);
    dim3 grid(M / 128, (N + 63) / 64);
    gemm_conv_wmma<<<grid, 256, 0, stream>>>(AB, WB, bias, out, M, N, K, M - 1, 1, 0, act, ldo);
  };

  // conv1d (512ch, same padding) + gelu, via multi-tap WMMA accumulation
  auto conv = [&](const float* A, const float* w, const float* bias, int ksz, int pad,
                  float* out) {
    cvt(A, AB, (size_t)NT * 512);
    size_t tot = (size_t)512 * 512 * ksz;
    repack_conv_bf16<<<(unsigned)((tot + 255) / 256), 256, 0, stream>>>(w, WB, 512, 512, ksz);
    dim3 grid(NT / 128, 8);
    gemm_conv_wmma<<<grid, 256, 0, stream>>>(AB, WB, bias, out, NT, 512, 512, L - 1,
                                             ksz, pad, 1, 512);
  };

  auto addk = [&](const float* a, const float* b, float* out, size_t n) {
    add_kernel<<<(unsigned)((n + 255) / 256), 256, 0, stream>>>(a, b, out, n);
  };
  auto affine = [&](const float* a, const float* g, const float* b, float* out, int act) {
    size_t n = (size_t)NT * 512;
    affine_kernel<<<(unsigned)((n + 255) / 256), 256, 0, stream>>>(a, g, b, out, n, 512, act);
  };
  auto ln = [&](const float* a, const float* res, const float* w, const float* b,
                float* out, int rows) {
    ln_kernel<<<rows, 256, 0, stream>>>(a, res, w, b, out, 512);
  };

  // Mamba2 block: in_proj -> dwconv+silu -> scan -> gated rmsnorm -> out_proj
  auto mamba = [&](const float* xin, int Btok, int nb, const float* const* p,
                   int dstate, int P, int rev, float* out, int ldo) {
    int H = 1024 / P;
    int cdim = 1024 + 2 * dstate;
    int din = 2048 + 2 * dstate + H;
    int M = nb * Btok;
    cvt(xin, AB, (size_t)M * 512);
    cvt(p[0], WB, (size_t)din * 512);
    dim3 g1(M / 128, (din + 63) / 64);
    gemm_conv_wmma<<<g1, 256, 0, stream>>>(AB, WB, nullptr, ZX, M, din, 512, M - 1,
                                           1, 0, 0, din);
    size_t tot = (size_t)M * cdim;
    dwconv_silu_kernel<<<(unsigned)((tot + 255) / 256), 256, 0, stream>>>(
        ZX, din, 1024, p[1], p[2], XC, cdim, Btok, nb, rev);
    if (dstate == 16)
      scan_kernel<16><<<nb * H, P, 0, stream>>>(ZX, din, 1024 + cdim, XC, cdim,
                                                p[3], p[4], p[5], YS, 1024, Btok, H, rev);
    else
      scan_kernel<64><<<nb * H, P, 0, stream>>>(ZX, din, 1024 + cdim, XC, cdim,
                                                p[3], p[4], p[5], YS, 1024, Btok, H, rev);
    gate_rms_kernel<<<M, 256, 0, stream>>>(YS, ZX, din, p[6], GT);
    cvt(GT, AB, (size_t)M * 1024);
    cvt(p[7], WB, (size_t)512 * 1024);
    dim3 g2(M / 128, 8);
    gemm_conv_wmma<<<g2, 256, 0, stream>>>(AB, WB, nullptr, out, M, 512, 1024, M - 1,
                                           1, 0, 0, ldo);
  };

  // Convblock (18 params: w1 b1 w2 b2 w3 b3 wl1 bl1 wl2 bl2 wf bf wr br bn1g bn1b bn2g bn2b)
  auto convblock = [&](const float* I, const float* const* p, float* O) {
    conv(I,  p[0],  p[1],  3, 1, T0);   // xa
    conv(T0, p[2],  p[3],  5, 2, T1);
    conv(T1, p[4],  p[5],  7, 3, T2);   // xb
    conv(T0, p[6],  p[7],  7, 3, T1);
    conv(T1, p[8],  p[9], 15, 7, T0);   // xc
    addk(T2, T0, T2, (size_t)NT * 512); // xb + xc
    conv(T2, p[10], p[11], 3, 1, T0);   // wf
    affine(T0, p[14], p[15], T0, 0);    // bn1
    conv(T0, p[12], p[13], 1, 0, T1);   // wr (1x1)
    affine(T1, p[16], p[17], O, 1);     // gelu(bn2)
  };

  const float* X = in[0];

  // ----------------- bimamba -----------------
  ln(X, nullptr, in[1], in[2], CUR, NT);                 // xn
  mamba(CUR, L, B, &in[5],  16, 64, 0, BI,       1024);  // fwd -> BI[:, :512]
  mamba(CUR, L, B, &in[13], 16, 64, 1, BI + 512, 1024);  // bwd -> BI[:, 512:]
  gemm(BI,  in[21], in[22], HID, NT, 2048, 1024, 1, 2048);  // gelu(bi@w1+b1)
  gemm(HID, in[23], in[24], NXT, NT, 512,  2048, 1, 512);   // gelu(.@w2+b2)
  ln(X, NXT, in[3], in[4], XA0, NT);                     // LN(x + ffn)

  // ----------------- conv blocks + dense -----------------
  convblock(XA0, &in[25], CUR);                          // conv1
  convblock(CUR, &in[43], SKIP);                         // conv2 -> skip
  gemm(CUR, in[115], in[116], NXT, NT, 512, 512, 1, 512); // gelu dense
  convblock(NXT, &in[61], CUR);                          // conv3

  // ----------------- stackmamba(local=CUR, glob=XA0) -----------------
  ln(CUR, nullptr, in[97], in[98], T0, NT);              // ln_l
  ln(XA0, nullptr, in[99], in[100], T1, NT);             // ln_g
  {
    size_t n = (size_t)NT * 512;
    copy_block_kernel<<<(unsigned)((n + 255) / 256), 256, 0, stream>>>(
        T1, CAT, L, 512, B, L, 2 * L);                   // ln_g first
    copy_block_kernel<<<(unsigned)((n + 255) / 256), 256, 0, stream>>>(
        T0, CAT + (size_t)L * 512, L, 512, B, L, 2 * L); // then ln_l
  }
  mamba(CAT, 2 * L, B, &in[103], 64, 128, 0, CAT, 512);  // out_proj overwrites CAT
  {
    size_t n = (size_t)NT * 512;
    copy_block_kernel<<<(unsigned)((n + 255) / 256), 256, 0, stream>>>(
        CAT + (size_t)L * 512, T2, L, 512, B, 2 * L, L); // x[:, -L:, :]
  }
  ln(T2, T0, in[101], in[102], T1, NT);                  // ln3(slice + ln_l)
  gemm(T1, in[111], in[112], T0, NT, 256, 512, 1, 256);  // gelu(x@w1+b1)
  gemm(T0, in[113], in[114], T2, NT, 512, 256, 0, 512);  // .@w2+b2
  addk(T2, T1, NXT, (size_t)NT * 512);                   // + x

  // ----------------- conv4 + concat -----------------
  convblock(NXT, &in[79], CUR);
  {
    size_t n = (size_t)NT * 1024;
    concat_kernel<<<(unsigned)((n + 255) / 256), 256, 0, stream>>>(
        SKIP, CUR, (float*)d_out, (size_t)NT);
  }
}
